// DeformableTransformer_85933705658995
// MI455X (gfx1250) — matrix operand order repaired and checked by reading, not compiled, not run on hardware
//
#include <hip/hip_runtime.h>
#include <hip/hip_bf16.h>

typedef __attribute__((ext_vector_type(16))) __bf16 v16bf;
typedef __attribute__((ext_vector_type(8)))  float  v8f;

#define BATCH   8
#define DMODEL  256
#define NHEAD   8
#define NLVL    4
#define NPTS    4
#define LTOT    3840
#define MTOT    (BATCH * LTOT)   // 30720
#define FFDIM   1024

// ---------------------------------------------------------------------------
// prep: vr[b][li] = mean(mask_li[b])
// ---------------------------------------------------------------------------
__global__ __launch_bounds__(256) void compute_vr_kernel(
    const float* __restrict__ m0, const float* __restrict__ m1,
    const float* __restrict__ m2, const float* __restrict__ m3,
    float* __restrict__ vr) {
  __shared__ float red[256];
  int blk = blockIdx.x;            // b*4 + li
  int b = blk >> 2, li = blk & 3;
  const float* mp; int T;
  if      (li == 0) { mp = m0; T = 2048; }
  else if (li == 1) { mp = m1; T = 1024; }
  else if (li == 2) { mp = m2; T = 512;  }
  else              { mp = m3; T = 256;  }
  const float* row = mp + (long)b * T;
  float s = 0.f;
  for (int i = threadIdx.x; i < T; i += 256) s += row[i];
  red[threadIdx.x] = s;
  __syncthreads();
  for (int st = 128; st > 0; st >>= 1) {
    if (threadIdx.x < st) red[threadIdx.x] += red[threadIdx.x + st];
    __syncthreads();
  }
  if (threadIdx.x == 0) vr[blk] = red[0] / (float)T;
}

// ---------------------------------------------------------------------------
// prep: x[b,l,d] = src[b,d,t];  pos[b,l,d] = posIn[b,d,t] + level_embed[jl,d]
// ---------------------------------------------------------------------------
__global__ __launch_bounds__(256) void build_xpos_kernel(
    const float* __restrict__ s0, const float* __restrict__ s1,
    const float* __restrict__ s2, const float* __restrict__ s3,
    const float* __restrict__ p0, const float* __restrict__ p1,
    const float* __restrict__ p2, const float* __restrict__ p3,
    const float* __restrict__ lev,
    float* __restrict__ x, float* __restrict__ pos) {
  long idx = (long)blockIdx.x * 256 + threadIdx.x;  // over B*L*D
  int d  = (int)(idx & 255);
  long bl = idx >> 8;
  int l = (int)(bl % LTOT);
  int b = (int)(bl / LTOT);
  const float *sp, *pp; int T, t, jl;
  if      (l < 2048) { jl = 0; t = l;        T = 2048; sp = s0; pp = p0; }
  else if (l < 3072) { jl = 1; t = l - 2048; T = 1024; sp = s1; pp = p1; }
  else if (l < 3584) { jl = 2; t = l - 3072; T = 512;  sp = s2; pp = p2; }
  else               { jl = 3; t = l - 3584; T = 256;  sp = s3; pp = p3; }
  long src_i = ((long)b * DMODEL + d) * T + t;
  x[idx]   = sp[src_i];
  pos[idx] = pp[src_i] + lev[jl * DMODEL + d];
}

// ---------------------------------------------------------------------------
// WMMA GEMM: C(M x N) = A(M x K) @ W(N x K)^T + bias  [optional relu/bf16-out]
// Block: 128(M) x 64(N); 8 waves, wave w -> rows [w*16, w*16+16).
// K stepped by 64 through ping-pong LDS; register-staged b128 global loads.
// ---------------------------------------------------------------------------
template<int ABF16, int ADDPOS, int RELU, int OUTBF16>
__global__ __launch_bounds__(256) void gemm_wmma_kernel(
    const void* __restrict__ Aptr, const float* __restrict__ Apos,
    const float* __restrict__ W, const float* __restrict__ bias,
    void* __restrict__ Cptr, int N, int K) {
  __shared__ __align__(64) __bf16 As[2][128 * 64];
  __shared__ __align__(64) __bf16 Bs[2][64 * 64];
  const int tid     = threadIdx.x;
  const int rowBase = blockIdx.x * 128;
  const int colBase = blockIdx.y * 64;
  const int wave = tid >> 5;
  const int lane = tid & 31;
  const int h    = lane >> 4;   // half-wave
  const int ln   = lane & 15;

  v8f acc[4];
#pragma unroll
  for (int nt = 0; nt < 4; ++nt)
#pragma unroll
    for (int j = 0; j < 8; ++j) acc[nt][j] = 0.f;

  const float*  Af = (const float*)Aptr;
  const __bf16* Ab = (const __bf16*)Aptr;

  // register staging: A tile 128x64 (8 chunks of 4/thread), B tile 64x64 (4 chunks)
  float4 ra[8], rp[8], rb[4];
  uint2  rab[8];

  auto loadTile = [&](int k0) {
#pragma unroll
    for (int i = 0; i < 8; ++i) {
      int idx = i * 256 + tid;
      int r = idx >> 4, kc = idx & 15;     // 16 lanes cover one row's 64 k
      long g = (long)(rowBase + r) * K + k0 + kc * 4;
      if (ABF16) rab[i] = *(const uint2*)(Ab + g);
      else {
        ra[i] = *(const float4*)(Af + g);
        if (ADDPOS) rp[i] = *(const float4*)(Apos + g);
      }
    }
#pragma unroll
    for (int i = 0; i < 4; ++i) {
      int idx = i * 256 + tid;
      int n = idx >> 4, kc = idx & 15;
      rb[i] = *(const float4*)(W + (long)(colBase + n) * K + k0 + kc * 4);
    }
  };
  auto storeTile = [&](int buf) {
#pragma unroll
    for (int i = 0; i < 8; ++i) {
      int idx = i * 256 + tid;
      int r = idx >> 4, kc = idx & 15;
      __bf16* dst = &As[buf][r * 64 + kc * 4];
      if (ABF16) { *(uint2*)dst = rab[i]; }
      else {
        float4 v = ra[i];
        if (ADDPOS) { v.x += rp[i].x; v.y += rp[i].y; v.z += rp[i].z; v.w += rp[i].w; }
        dst[0] = (__bf16)v.x; dst[1] = (__bf16)v.y;
        dst[2] = (__bf16)v.z; dst[3] = (__bf16)v.w;
      }
    }
#pragma unroll
    for (int i = 0; i < 4; ++i) {
      int idx = i * 256 + tid;
      int n = idx >> 4, kc = idx & 15;
      __bf16* dst = &Bs[buf][n * 64 + kc * 4];
      float4 v = rb[i];
      dst[0] = (__bf16)v.x; dst[1] = (__bf16)v.y;
      dst[2] = (__bf16)v.z; dst[3] = (__bf16)v.w;
    }
  };

  loadTile(0);
  storeTile(0);
  __syncthreads();

  const int nk = K >> 6;   // K/64
  for (int kk = 0; kk < nk; ++kk) {
    const int cur = kk & 1;
    if (kk + 1 < nk) loadTile((kk + 1) * 64);   // overlap with compute below

    union F16 { v16bf v; __bf16 e[16]; };
#pragma unroll
    for (int ks = 0; ks < 2; ++ks) {
      // A fragment: row = wave*16 + ln; halves 0-7 -> K 8h.., 8-15 -> K 16+8h..
      F16 a;
      const __bf16* arow = &As[cur][(wave * 16 + ln) * 64 + ks * 32];
#pragma unroll
      for (int j = 0; j < 8; ++j) {
        a.e[j]     = arow[8 * h + j];
        a.e[8 + j] = arow[16 + 8 * h + j];
      }
#pragma unroll
      for (int nt = 0; nt < 4; ++nt) {
        // B fragment: col = nt*16 + ln; halves 0-7 -> K 8h.., 8-15 -> K 16+8h..
        F16 bfr;
        const __bf16* brow = &Bs[cur][(nt * 16 + ln) * 64 + ks * 32];
#pragma unroll
        for (int j = 0; j < 8; ++j) { bfr.e[j] = brow[8 * h + j]; bfr.e[8 + j] = brow[16 + 8 * h + j]; }
        acc[nt] = __builtin_amdgcn_wmma_f32_16x16x32_bf16(
            false, a.v, false, bfr.v, (short)0, acc[nt], false, false);
      }
    }
    if (kk + 1 < nk) storeTile(1 - cur);
    __syncthreads();
  }

  float*  Cf = (float*)Cptr;
  __bf16* Cb = (__bf16*)Cptr;
#pragma unroll
  for (int nt = 0; nt < 4; ++nt) {
    int col = colBase + nt * 16 + ln;
    float bv = bias[col];
#pragma unroll
    for (int v = 0; v < 8; ++v) {
      int row = rowBase + wave * 16 + v + 8 * h;   // C layout: VGPR v -> M = v + 8h
      float val = acc[nt][v] + bv;
      if (RELU) val = val > 0.f ? val : 0.f;
      if (OUTBF16) Cb[(long)row * N + col] = (__bf16)val;
      else         Cf[(long)row * N + col] = val;
    }
  }
}

// ---------------------------------------------------------------------------
// softmax over 16 contiguous attn-weight logits per (b,l,h)
// ---------------------------------------------------------------------------
__global__ __launch_bounds__(256) void softmax16_kernel(float* __restrict__ aw, int n) {
  int i = blockIdx.x * 256 + threadIdx.x;
  if (i >= n) return;
  float* p = aw + (long)i * 16;
  float v[16], mx = -1e30f;
#pragma unroll
  for (int j = 0; j < 16; ++j) { v[j] = p[j]; mx = v[j] > mx ? v[j] : mx; }
  float s = 0.f;
#pragma unroll
  for (int j = 0; j < 16; ++j) { v[j] = __expf(v[j] - mx); s += v[j]; }
  float inv = 1.f / s;
#pragma unroll
  for (int j = 0; j < 16; ++j) p[j] = v[j] * inv;
}

// ---------------------------------------------------------------------------
// deformable sampling: one wave per token; lane = channel within head (Dh=32)
// ---------------------------------------------------------------------------
__global__ __launch_bounds__(256) void deform_sample_kernel(
    const float* __restrict__ value, const float* __restrict__ off,
    const float* __restrict__ aw, const float* __restrict__ vr,
    float* __restrict__ attn) {
  int gwave = (blockIdx.x * 256 + threadIdx.x) >> 5;   // b*L + l
  int lane  = threadIdx.x & 31;
  if (gwave >= MTOT) return;
  int b = gwave / LTOT, l = gwave % LTOT;
  const int lens[4]   = {2048, 1024, 512, 256};
  const int starts[4] = {0, 2048, 3072, 3584};
  int jl, t;
  if      (l < 2048) { jl = 0; t = l; }
  else if (l < 3072) { jl = 1; t = l - 2048; }
  else if (l < 3584) { jl = 2; t = l - 3072; }
  else               { jl = 3; t = l - 3584; }
  float ref1 = (t + 0.5f) / (float)lens[jl];
  long tokOff = (long)gwave * (NHEAD * NLVL * NPTS);
#pragma unroll
  for (int hh = 0; hh < NHEAD; ++hh) {
    float acc = 0.f;
#pragma unroll
    for (int li = 0; li < NLVL; ++li) {
      float refv = ref1 * vr[b * NLVL + li];
      int T = lens[li], start = starts[li];
      float Tf = (float)T;
#pragma unroll
      for (int p = 0; p < NPTS; ++p) {
        long o  = tokOff + hh * 16 + li * 4 + p;
        float of = off[o];
        float a_w = aw[o];
        float loc = refv + of / Tf;
        float pc  = loc * Tf - 0.5f;
        float fl  = floorf(pc);
        int i0 = (int)fl, i1 = i0 + 1;
        float w1 = pc - fl;
        float m0 = (i0 >= 0 && i0 < T) ? 1.f : 0.f;
        float m1 = (i1 >= 0 && i1 < T) ? 1.f : 0.f;
        int c0 = i0 < 0 ? 0 : (i0 > T - 1 ? T - 1 : i0);
        int c1 = i1 < 0 ? 0 : (i1 > T - 1 ? T - 1 : i1);
        float v0 = value[((long)b * LTOT + start + c0) * DMODEL + hh * 32 + lane] * m0;
        float v1 = value[((long)b * LTOT + start + c1) * DMODEL + hh * 32 + lane] * m1;
        acc += a_w * (v0 * (1.f - w1) + v1 * w1);
      }
    }
    attn[(long)gwave * DMODEL + hh * 32 + lane] = acc;
  }
}

// ---------------------------------------------------------------------------
// x = LayerNorm(x + a) * g + be   (wave per row, D=256 -> 8 per lane)
// ---------------------------------------------------------------------------
__global__ __launch_bounds__(256) void add_ln_kernel(
    float* __restrict__ x, const float* __restrict__ a,
    const float* __restrict__ g, const float* __restrict__ be) {
  int row  = (blockIdx.x * 256 + threadIdx.x) >> 5;
  int lane = threadIdx.x & 31;
  if (row >= MTOT) return;
  long base = (long)row * DMODEL;
  float v[8], s = 0.f;
#pragma unroll
  for (int j = 0; j < 8; ++j) {
    int c = j * 32 + lane;
    v[j] = x[base + c] + a[base + c];
    s += v[j];
  }
#pragma unroll
  for (int m = 16; m >= 1; m >>= 1) s += __shfl_xor(s, m, 32);
  float mean = s * (1.f / DMODEL);
  float var = 0.f;
#pragma unroll
  for (int j = 0; j < 8; ++j) { float d = v[j] - mean; var += d * d; }
#pragma unroll
  for (int m = 16; m >= 1; m >>= 1) var += __shfl_xor(var, m, 32);
  var *= (1.f / DMODEL);
  float r = rsqrtf(var + 1e-5f);
#pragma unroll
  for (int j = 0; j < 8; ++j) {
    int c = j * 32 + lane;
    x[base + c] = (v[j] - mean) * r * g[c] + be[c];
  }
}

// ---------------------------------------------------------------------------
extern "C" void kernel_launch(void* const* d_in, const int* in_sizes, int n_in,
                              void* d_out, int out_size, void* d_ws, size_t ws_size,
                              hipStream_t stream) {
  (void)in_sizes; (void)n_in; (void)out_size; (void)ws_size;
  const float* src[4]  = {(const float*)d_in[0], (const float*)d_in[3],
                          (const float*)d_in[6], (const float*)d_in[9]};
  const float* posi[4] = {(const float*)d_in[1], (const float*)d_in[4],
                          (const float*)d_in[7], (const float*)d_in[10]};
  const float* mask[4] = {(const float*)d_in[2], (const float*)d_in[5],
                          (const float*)d_in[8], (const float*)d_in[11]};
  const float* lev   = (const float*)d_in[12];
  const float* W_off = (const float*)d_in[13];
  const float* b_off = (const float*)d_in[14];
  const float* W_aw  = (const float*)d_in[15];
  const float* b_aw  = (const float*)d_in[16];
  const float* W_val = (const float*)d_in[17];
  const float* b_val = (const float*)d_in[18];
  const float* W_out = (const float*)d_in[19];
  const float* b_out = (const float*)d_in[20];
  const float* g1    = (const float*)d_in[21];
  const float* be1   = (const float*)d_in[22];
  const float* W1    = (const float*)d_in[23];
  const float* b1    = (const float*)d_in[24];
  const float* W2    = (const float*)d_in[25];
  const float* b2    = (const float*)d_in[26];
  const float* g2    = (const float*)d_in[27];
  const float* be2   = (const float*)d_in[28];

  float* x = (float*)d_out;                        // resident activation (B*L, D)
  char* ws = (char*)d_ws;
  const long MD  = (long)MTOT * DMODEL;            // 7,864,320
  const long MHP = (long)MTOT * 128;               // 3,932,160
  float*  pos   = (float*)(ws);
  float*  value = (float*)(ws + MD  * 4);          // also reused as GEMM tmp
  float*  offb  = (float*)(ws + MD  * 8);
  float*  awb   = (float*)(ws + MD  * 8 + MHP * 4);
  float*  attn  = (float*)(ws + MD  * 8 + MHP * 8);
  __bf16* h1    = (__bf16*)(ws + MD * 12 + MHP * 8);
  float*  vr    = (float*)(ws + MD * 12 + MHP * 8 + (long)MTOT * FFDIM * 2);

  compute_vr_kernel<<<32, 256, 0, stream>>>(mask[0], mask[1], mask[2], mask[3], vr);
  build_xpos_kernel<<<(int)(MD / 256), 256, 0, stream>>>(
      src[0], src[1], src[2], src[3], posi[0], posi[1], posi[2], posi[3],
      lev, x, pos);

  const int GM = MTOT / 128;  // 240
  for (int l = 0; l < 6; ++l) {
    const float* Wv = W_val + (long)l * 256 * 256; const float* bv = b_val + l * 256;
    const float* Wo = W_off + (long)l * 128 * 256; const float* bo = b_off + l * 128;
    const float* Wa = W_aw  + (long)l * 128 * 256; const float* ba = b_aw  + l * 128;
    const float* Wq = W_out + (long)l * 256 * 256; const float* bq = b_out + l * 256;
    const float* Wf1 = W1 + (long)l * FFDIM * 256; const float* bf1 = b1 + l * FFDIM;
    const float* Wf2 = W2 + (long)l * 256 * FFDIM; const float* bf2 = b2 + l * 256;

    // value = x @ Wv^T + bv
    gemm_wmma_kernel<0,0,0,0><<<dim3(GM, 4), 256, 0, stream>>>(
        x, nullptr, Wv, bv, value, 256, 256);
    // off = (x+pos) @ Woff^T + boff ; aw logits likewise
    gemm_wmma_kernel<0,1,0,0><<<dim3(GM, 2), 256, 0, stream>>>(
        x, pos, Wo, bo, offb, 128, 256);
    gemm_wmma_kernel<0,1,0,0><<<dim3(GM, 2), 256, 0, stream>>>(
        x, pos, Wa, ba, awb, 128, 256);
    softmax16_kernel<<<(MTOT * NHEAD) / 256, 256, 0, stream>>>(awb, MTOT * NHEAD);
    deform_sample_kernel<<<MTOT / 8, 256, 0, stream>>>(value, offb, awb, vr, attn);
    // a = attn @ Wout^T + bout   (value buffer reused as tmp)
    gemm_wmma_kernel<0,0,0,0><<<dim3(GM, 4), 256, 0, stream>>>(
        attn, nullptr, Wq, bq, value, 256, 256);
    add_ln_kernel<<<MTOT / 8, 256, 0, stream>>>(x, value, g1 + l * 256, be1 + l * 256);
    // FFN
    gemm_wmma_kernel<0,0,1,1><<<dim3(GM, 16), 256, 0, stream>>>(
        x, nullptr, Wf1, bf1, h1, FFDIM, 256);
    gemm_wmma_kernel<1,0,0,0><<<dim3(GM, 4), 256, 0, stream>>>(
        h1, nullptr, Wf2, bf2, value, 256, FFDIM);
    add_ln_kernel<<<MTOT / 8, 256, 0, stream>>>(x, value, g2 + l * 256, be2 + l * 256);
  }
}
